// Multi_Head_Attention_16844861735086
// MI455X (gfx1250) — compile-verified, matrix operand order checked by reading, not run on hardware
//
#include <hip/hip_runtime.h>
#include <cstdint>

typedef __attribute__((ext_vector_type(16))) __bf16 v16bf;
typedef __attribute__((ext_vector_type(8)))  float  v8f;
typedef __attribute__((__vector_size__(16))) int    v4i_n;   // native 16B vector

#define B_    2
#define S_    2048
#define H_    16
#define HD_   64
#define DM_   1024
#define NQKV_ 3072
#define M_    (B_ * S_)      // 4096

#define AS1 __attribute__((address_space(1)))
#define AS3 __attribute__((address_space(3)))

// ---- CDNA5 async global->LDS staging (ASYNCcnt-tracked), with fallbacks ----
#if __has_builtin(__builtin_amdgcn_global_load_async_to_lds_b128)
#define HAVE_ASYNC_LDS 1
#else
#define HAVE_ASYNC_LDS 0
#endif

#if HAVE_ASYNC_LDS
#if __has_builtin(__builtin_amdgcn_s_wait_asynccnt)
#define WAIT_ASYNC(n) __builtin_amdgcn_s_wait_asynccnt(n)
#else
#define WAIT_ASYNC(n) asm volatile("s_wait_asynccnt %0" ::"n"(n))
#endif
#else
#define WAIT_ASYNC(n) ((void)0)
#endif

// Split workgroup barrier WITHOUT the __syncthreads() memory-fence flush
// (no s_wait_loadcnt 0): ordering of async-LDS writes is handled explicitly
// with s_wait_asynccnt, and ds-read completion is enforced by the data
// dependences (s_wait_dscnt before the consuming WMMAs). "memory" clobber
// keeps the compiler from reordering memory ops across the barrier.
__device__ inline void wg_barrier() {
#if HAVE_ASYNC_LDS
    asm volatile("s_barrier_signal -1\n\ts_barrier_wait -1" ::: "memory");
#else
    __syncthreads();
#endif
}

__device__ inline void stage16(__bf16* ldst, const __bf16* gsrc) {
#if HAVE_ASYNC_LDS
    __builtin_amdgcn_global_load_async_to_lds_b128((AS1 v4i_n*)gsrc,
                                                   (AS3 v4i_n*)ldst, 0, 0);
#else
    *(int4*)ldst = *(const int4*)gsrc;   // sync fallback: ds_store_b128
#endif
}

__device__ inline v8f wmma_bf16(v16bf a, v16bf b, v8f c) {
    return __builtin_amdgcn_wmma_f32_16x16x32_bf16(false, a, false, b, (short)0, c,
                                                   false, false);
}

// Load a 16x32 bf16 operand tile (A layout; B-operand for A@B^T reads identically).
// lanes 0-15: VGPR j<4 -> K=2j,2j+1 ; j>=4 -> K=16+2(j-4).. ; lanes 16-31: +8 / +24.
__device__ inline v16bf load_tile(const __bf16* base, int row_stride) {
    int lane = threadIdx.x & 31;
    int r    = lane & 15;
    int koff = (lane >> 4) << 3;
    const __bf16* p = base + (size_t)r * row_stride;
    v16bf a;
#pragma unroll
    for (int j = 0; j < 8; ++j) {
        int k = (j < 4) ? (koff + 2 * j) : (16 + koff + 2 * (j - 4));
        a[2 * j]     = p[k];
        a[2 * j + 1] = p[k + 1];
    }
    return a;
}

// ---------------- elementwise converts ----------------
__global__ __launch_bounds__(256) void f32_to_bf16_k(const float* __restrict__ in,
                                                     __bf16* __restrict__ out, int n) {
    int i = blockIdx.x * 256 + threadIdx.x;
    if (i < n) out[i] = (__bf16)in[i];
}

// in is [K][N] f32, out is [N][K] bf16
__global__ __launch_bounds__(256) void transpose_to_bf16_k(const float* __restrict__ in,
                                                           __bf16* __restrict__ out,
                                                           int K, int N) {
    int i = blockIdx.x * 256 + threadIdx.x;
    if (i >= K * N) return;
    int n = i / K, k = i % K;
    out[i] = (__bf16)in[(size_t)k * N + n];
}

// 8 WMMAs over one 32-K slab: load all four B tiles into distinct registers
// first so the ds_loads pipeline, then issue the WMMA block.
__device__ inline void slab_mma(const __bf16* slab, v16bf a0, v16bf a1, v8f acc[2][4]) {
    v16bf b0 = load_tile(slab + 0 * 512, 32);
    v16bf b1 = load_tile(slab + 1 * 512, 32);
    v16bf b2 = load_tile(slab + 2 * 512, 32);
    v16bf b3 = load_tile(slab + 3 * 512, 32);
    acc[0][0] = wmma_bf16(a0, b0, acc[0][0]);
    acc[1][0] = wmma_bf16(a1, b0, acc[1][0]);
    acc[0][1] = wmma_bf16(a0, b1, acc[0][1]);
    acc[1][1] = wmma_bf16(a1, b1, acc[1][1]);
    acc[0][2] = wmma_bf16(a0, b2, acc[0][2]);
    acc[1][2] = wmma_bf16(a1, b2, acc[1][2]);
    acc[0][3] = wmma_bf16(a0, b3, acc[0][3]);
    acc[1][3] = wmma_bf16(a1, b3, acc[1][3]);
}

// Shared inner pipeline for the two projection GEMMs.
// Block: 256 threads = 8 waves, 256(M) x 64(N); wave owns 32(M) x 64(N).
// B slab 64(N)x32(K) bf16 = 4KB async-staged to LDS, double buffered; K-loop
// unrolled x2 so A-register buffers and LDS slabs alternate without copies.
__device__ inline void gemm_core(const __bf16* arow0, const __bf16* arow1,
                                 const __bf16* srow, __bf16* ldsB, int sn, int sk,
                                 v8f acc[2][4]) {
    stage16(&ldsB[sn * 32 + sk], srow);                 // slab0 <- kc 0
    v16bf a0c = load_tile(arow0, DM_);
    v16bf a1c = load_tile(arow1, DM_);
    v16bf a0n, a1n;
    for (int kc = 0; kc < DM_; kc += 64) {
        bool last = (kc + 64) >= DM_;
        // ---- phase A: compute slab0(kc), prefetch slab1(kc+32) ----
        stage16(&ldsB[2048 + sn * 32 + sk], srow + kc + 32);
        a0n = load_tile(arow0 + kc + 32, DM_);
        a1n = load_tile(arow1 + kc + 32, DM_);
        WAIT_ASYNC(1);
        wg_barrier();
        slab_mma(ldsB, a0c, a1c, acc);
        wg_barrier();
        // ---- phase B: compute slab1(kc+32), prefetch slab0(kc+64) ----
        if (!last) {
            stage16(&ldsB[sn * 32 + sk], srow + kc + 64);
            a0c = load_tile(arow0 + kc + 64, DM_);
            a1c = load_tile(arow1 + kc + 64, DM_);
            WAIT_ASYNC(1);
        } else {
            WAIT_ASYNC(0);
        }
        wg_barrier();
        slab_mma(ldsB + 2048, a0n, a1n, acc);
        wg_barrier();
    }
}

// ---------------- QKV projection GEMM ----------------
__global__ __launch_bounds__(256) void qkv_gemm_k(const __bf16* __restrict__ Xb,
                                                  const __bf16* __restrict__ WT,
                                                  const float*  __restrict__ bias,
                                                  __bf16* __restrict__ Qb,
                                                  __bf16* __restrict__ Kb,
                                                  __bf16* __restrict__ Vt) {
    __shared__ __align__(16) __bf16 ldsB[2 * 64 * 32];
    int tid  = threadIdx.x;
    int lane = tid & 31;
    int w    = tid >> 5;
    int n0   = blockIdx.x * 64;            // 48 blocks
    int m0   = blockIdx.y * 256 + w * 32;  // 16 blocks, 8 waves x 32 rows

    int sn = tid >> 2;
    int sk = (tid & 3) * 8;
    const __bf16* srow = WT + (size_t)(n0 + sn) * DM_ + sk;

    v8f acc[2][4] = {};
    gemm_core(Xb + (size_t)m0 * DM_, Xb + (size_t)(m0 + 16) * DM_, srow, ldsB, sn, sk,
              acc);

    int half = lane >> 4, nc = lane & 15;
#pragma unroll
    for (int r = 0; r < 2; ++r) {
        int m0r = m0 + 16 * r;
#pragma unroll
        for (int t = 0; t < 4; ++t) {
            int n   = n0 + 16 * t + nc;
            int h   = n / (3 * HD_);
            int rem = n % (3 * HD_);
            int ty  = rem / HD_;
            int hd  = rem % HD_;
            float bv = bias[n];
#pragma unroll
            for (int i = 0; i < 8; ++i) {
                int m  = m0r + i + 8 * half;
                int bb = m >> 11;
                int s  = m & 2047;
                __bf16 v = (__bf16)(acc[r][t][i] + bv);
                if (ty == 0)
                    Qb[(((size_t)(bb * H_ + h) * S_) + s) * HD_ + hd] = v;
                else if (ty == 1)
                    Kb[(((size_t)(bb * H_ + h) * S_) + s) * HD_ + hd] = v;
                else
                    Vt[(((size_t)(bb * H_ + h) * HD_) + hd) * S_ + s] = v;
            }
        }
    }
}

// ---------------- flash attention ----------------
// One wave owns 16 q-rows of one (b,h). Streams 32-key tiles with online softmax.
__global__ __launch_bounds__(256) void attn_k(const __bf16* __restrict__ Qb,
                                              const __bf16* __restrict__ Kb,
                                              const __bf16* __restrict__ Vt,
                                              __bf16* __restrict__ Ob) {
    __shared__ __bf16 plds[8 * 16 * 32];      // per-wave 16x32 bf16 P tile
    const float scale = 0.125f;               // 1/sqrt(64)
    const float BIGN  = -3.0e38f;

    int lane = threadIdx.x & 31;
    int w    = threadIdx.x >> 5;
    int bh   = blockIdx.y;
    int b    = bh >> 4, h = bh & 15;
    int q0   = blockIdx.x * 128 + w * 16;     // wave-uniform

    const __bf16* Qbase = Qb + ((size_t)(b * H_ + h) * S_ + q0) * HD_;
    const __bf16* Kbase = Kb + (size_t)(b * H_ + h) * S_ * HD_;
    const __bf16* Vbase = Vt + (size_t)(b * H_ + h) * HD_ * S_;
    __bf16* lp = plds + w * (16 * 32);

    v16bf qa0 = load_tile(Qbase + 0, HD_);
    v16bf qa1 = load_tile(Qbase + 32, HD_);

    v8f acc0 = {}, acc1 = {}, acc2 = {}, acc3 = {};
    float mrun[8], lrun[8];
#pragma unroll
    for (int i = 0; i < 8; ++i) { mrun[i] = BIGN; lrun[i] = 0.0f; }

    int half = lane >> 4, nc = lane & 15;
    int kend = q0 + 16;                       // causal bound (wave-uniform)

    for (int kv = 0; kv < kend; kv += 32) {
        // ---- V operand tiles issued FIRST: their latency hides under softmax ----
        v16bf bv0 = load_tile(Vbase + kv +  0 * S_, S_);
        v16bf bv1 = load_tile(Vbase + kv + 16 * S_, S_);
        v16bf bv2 = load_tile(Vbase + kv + 32 * S_, S_);
        v16bf bv3 = load_tile(Vbase + kv + 48 * S_, S_);
        // ---- scores: two 16x16 tiles over 32 keys ----
        v8f s0 = {}, s1 = {};
        {
            v16bf bk;
            bk = load_tile(Kbase + (size_t)kv * HD_ + 0, HD_);         s0 = wmma_bf16(qa0, bk, s0);
            bk = load_tile(Kbase + (size_t)kv * HD_ + 32, HD_);        s0 = wmma_bf16(qa1, bk, s0);
            bk = load_tile(Kbase + (size_t)(kv + 16) * HD_ + 0, HD_);  s1 = wmma_bf16(qa0, bk, s1);
            bk = load_tile(Kbase + (size_t)(kv + 16) * HD_ + 32, HD_); s1 = wmma_bf16(qa1, bk, s1);
        }
        // ---- scale + causal mask + online softmax (fp32, C-layout) ----
        int k0g = kv + nc, k1g = kv + 16 + nc;
#pragma unroll
        for (int i = 0; i < 8; ++i) {
            int row = q0 + i + 8 * half;
            float a0 = (k0g <= row) ? s0[i] * scale : BIGN;
            float a1 = (k1g <= row) ? s1[i] * scale : BIGN;
            s0[i] = a0; s1[i] = a1;
            float mx = fmaxf(a0, a1);
            mx = fmaxf(mx, __shfl_xor(mx, 1, 32));
            mx = fmaxf(mx, __shfl_xor(mx, 2, 32));
            mx = fmaxf(mx, __shfl_xor(mx, 4, 32));
            mx = fmaxf(mx, __shfl_xor(mx, 8, 32));   // row-max within 16-lane half
            float mo = mrun[i];
            float m  = fmaxf(mo, mx);
            float alpha = __expf(mo - m);
            float p0 = __expf(s0[i] - m);
            float p1 = __expf(s1[i] - m);
            float rs = p0 + p1;
            rs += __shfl_xor(rs, 1, 32);
            rs += __shfl_xor(rs, 2, 32);
            rs += __shfl_xor(rs, 4, 32);
            rs += __shfl_xor(rs, 8, 32);
            lrun[i] = lrun[i] * alpha + rs;
            mrun[i] = m;
            acc0[i] *= alpha; acc1[i] *= alpha; acc2[i] *= alpha; acc3[i] *= alpha;
            // P -> LDS (row-major 16x32 bf16); wave-ordered DS ops, no barrier needed
            lp[(i + 8 * half) * 32 + nc]      = (__bf16)p0;
            lp[(i + 8 * half) * 32 + nc + 16] = (__bf16)p1;
        }
        // ---- O += P @ V  (P: 16x32 A-operand from LDS) ----
        v16bf pa = load_tile(lp, 32);
        acc0 = wmma_bf16(pa, bv0, acc0);
        acc1 = wmma_bf16(pa, bv1, acc1);
        acc2 = wmma_bf16(pa, bv2, acc2);
        acc3 = wmma_bf16(pa, bv3, acc3);
    }
    // ---- normalize + write concat layout [b][s][h*64+hd] ----
#pragma unroll
    for (int i = 0; i < 8; ++i) {
        float inv = 1.0f / lrun[i];
        int row = q0 + i + 8 * half;
        size_t base = ((size_t)b * S_ + row) * DM_ + h * HD_;
        Ob[base +  0 + nc] = (__bf16)(acc0[i] * inv);
        Ob[base + 16 + nc] = (__bf16)(acc1[i] * inv);
        Ob[base + 32 + nc] = (__bf16)(acc2[i] * inv);
        Ob[base + 48 + nc] = (__bf16)(acc3[i] * inv);
    }
}

// ---------------- output projection GEMM ----------------
__global__ __launch_bounds__(256) void out_gemm_k(const __bf16* __restrict__ Ob,
                                                  const __bf16* __restrict__ WT,
                                                  const float*  __restrict__ bias,
                                                  float* __restrict__ out) {
    __shared__ __align__(16) __bf16 ldsB[2 * 64 * 32];
    int tid  = threadIdx.x;
    int lane = tid & 31;
    int w    = tid >> 5;
    int n0   = blockIdx.x * 64;            // 16 blocks
    int m0   = blockIdx.y * 256 + w * 32;  // 16 blocks

    int sn = tid >> 2;
    int sk = (tid & 3) * 8;
    const __bf16* srow = WT + (size_t)(n0 + sn) * DM_ + sk;

    v8f acc[2][4] = {};
    gemm_core(Ob + (size_t)m0 * DM_, Ob + (size_t)(m0 + 16) * DM_, srow, ldsB, sn, sk,
              acc);

    int half = lane >> 4, nc = lane & 15;
#pragma unroll
    for (int r = 0; r < 2; ++r) {
        int m0r = m0 + 16 * r;
#pragma unroll
        for (int t = 0; t < 4; ++t) {
            int n = n0 + 16 * t + nc;
            float bv = bias[n];
#pragma unroll
            for (int i = 0; i < 8; ++i) {
                int m = m0r + i + 8 * half;
                out[(size_t)m * DM_ + n] = acc[r][t][i] + bv;
            }
        }
    }
}

extern "C" void kernel_launch(void* const* d_in, const int* in_sizes, int n_in,
                              void* d_out, int out_size, void* d_ws, size_t ws_size,
                              hipStream_t stream) {
    const float* x     = (const float*)d_in[0];
    const float* w_qkv = (const float*)d_in[1];
    const float* b_qkv = (const float*)d_in[2];
    const float* w_out = (const float*)d_in[3];
    const float* b_out = (const float*)d_in[4];
    float* out = (float*)d_out;

    char* ws = (char*)d_ws;
    __bf16* xb    = (__bf16*)ws; ws += (size_t)M_ * DM_ * 2;           // 8 MB
    __bf16* wqkvT = (__bf16*)ws; ws += (size_t)NQKV_ * DM_ * 2;        // 6 MB
    __bf16* woutT = (__bf16*)ws; ws += (size_t)DM_ * DM_ * 2;          // 2 MB
    __bf16* Qb    = (__bf16*)ws; ws += (size_t)B_ * H_ * S_ * HD_ * 2; // 8 MB
    __bf16* Kb    = (__bf16*)ws; ws += (size_t)B_ * H_ * S_ * HD_ * 2; // 8 MB
    __bf16* Vt    = (__bf16*)ws; ws += (size_t)B_ * H_ * S_ * HD_ * 2; // 8 MB
    __bf16* Ob    = (__bf16*)ws; ws += (size_t)M_ * DM_ * 2;           // 8 MB

    int nx = M_ * DM_;
    f32_to_bf16_k<<<(nx + 255) / 256, 256, 0, stream>>>(x, xb, nx);
    int nw = DM_ * NQKV_;
    transpose_to_bf16_k<<<(nw + 255) / 256, 256, 0, stream>>>(w_qkv, wqkvT, DM_, NQKV_);
    int no = DM_ * DM_;
    transpose_to_bf16_k<<<(no + 255) / 256, 256, 0, stream>>>(w_out, woutT, DM_, DM_);

    qkv_gemm_k<<<dim3(NQKV_ / 64, M_ / 256), 256, 0, stream>>>(xb, wqkvT, b_qkv,
                                                               Qb, Kb, Vt);
    attn_k<<<dim3(S_ / 128, B_ * H_), 256, 0, stream>>>(Qb, Kb, Vt, Ob);
    out_gemm_k<<<dim3(DM_ / 64, M_ / 256), 256, 0, stream>>>(Ob, woutT, b_out, out);
}
